// PushupLSTM_46437186404649
// MI455X (gfx1250) — matrix operand order repaired and checked by reading, not compiled, not run on hardware
//
#include <hip/hip_runtime.h>
#include <hip/hip_bf16.h>
#include <cstdint>

typedef __bf16 bf16_t;
typedef __attribute__((ext_vector_type(16))) __bf16 v16bf;
typedef __attribute__((ext_vector_type(8)))  __bf16 v8bf;
typedef __attribute__((ext_vector_type(8)))  float  v8f;

#define HIDDEN   128
#define GATES    512           // 4*HIDDEN
#define BATCH    256
#define TSTEPS   2048
#define ROWS_PB  16            // batch rows per block
#define BLK_THR  1024          // 32 waves

// ---- workspace layout (bytes) ----
#define WS_W0     0                          // bf16 [512][128]   W_hh0
#define WS_WC1    (WS_W0  + 512*128*2)       // bf16 [512][256]   [W_ih1 | W_hh1]
#define WS_B0     (WS_WC1 + 512*256*2)       // f32  [512]        b_ih0+b_hh0
#define WS_B1     (WS_B0  + 512*4)           // f32  [512]
#define WS_HFIN   (WS_B1  + 512*4)           // f32  [256][128]   final h of layer1
#define WS_H1SEQ  (WS_HFIN + 256*128*4)      // bf16 [T][256][128] layer0 output seq

// ------------------------------------------------------------------
// Prep: fp32 weights -> bf16 (row-major, matches WMMA B-fragment
// layout), fuse bias vectors.
// ------------------------------------------------------------------
__global__ void prep_kernel(const float* __restrict__ W_hh0,
                            const float* __restrict__ W_ih1,
                            const float* __restrict__ W_hh1,
                            const float* __restrict__ b_ih0,
                            const float* __restrict__ b_hh0,
                            const float* __restrict__ b_ih1,
                            const float* __restrict__ b_hh1,
                            bf16_t* __restrict__ W0,
                            bf16_t* __restrict__ Wc1,
                            float* __restrict__ b0,
                            float* __restrict__ b1) {
  int n = blockIdx.x;        // 0..511 gate row
  int k = threadIdx.x;       // 0..127
  W0 [n*HIDDEN + k]        = (bf16_t)W_hh0[n*HIDDEN + k];
  Wc1[n*256 + k]           = (bf16_t)W_ih1[n*HIDDEN + k];
  Wc1[n*256 + HIDDEN + k]  = (bf16_t)W_hh1[n*HIDDEN + k];
  if (k == 0) {
    b0[n] = b_ih0[n] + b_hh0[n];
    b1[n] = b_ih1[n] + b_hh1[n];
  }
}

// Branch-free activations: v_exp_f32 + v_rcp_f32, saturate correctly
// at +/-inf (exp2(+inf)->inf, rcp(inf)->0; exp2(-inf)->0, rcp(1)->1).
__device__ __forceinline__ float fsig(float v) {
  return __builtin_amdgcn_rcpf(1.0f +
         __builtin_amdgcn_exp2f(v * -1.4426950408889634f));
}
__device__ __forceinline__ float ftanh(float v) {
  return 2.0f * __builtin_amdgcn_rcpf(1.0f +
         __builtin_amdgcn_exp2f(v * -2.8853900817779268f)) - 1.0f;
}

// ------------------------------------------------------------------
// One LSTM layer, whole sequence. K = A-matrix depth:
//   K=128 : layer 0, A = h_{t-1};  x-projection (K=5) done scalar;
//           h sequence written out via global_store_async_from_lds.
//   K=256 : layer 1, A = [x_t | h_{t-1}], x_t async-prefetched to LDS.
// Block: 16 batch rows, 1024 threads = 32 waves, wave w owns gate
// columns [16w,16w+16). Weights live in VGPRs for the whole t-loop.
// ------------------------------------------------------------------
template<int K>
__global__ __launch_bounds__(BLK_THR)
void lstm_layer_kernel(const float*  __restrict__ x,         // [B][T][5]   (K==128)
                       const float*  __restrict__ W_ih0,     // [512][5]    (K==128)
                       const bf16_t* __restrict__ Wb,        // bf16 [512][K]
                       const float*  __restrict__ bias,      // [512]
                       const bf16_t* __restrict__ h1seq_in,  // [T][B][128] (K==256)
                       bf16_t*       __restrict__ h1seq_out, // [T][B][128] (K==128)
                       float*        __restrict__ hfinal)    // [B][128]    (K==256)
{
  __shared__ bf16_t a_lds[ROWS_PB][K];        // A operand: [x |] h  (bf16)
  __shared__ float  gates_lds[ROWS_PB][GATES];// pre-activation gates (f32)
  __shared__ float  x_lds[ROWS_PB][5];        // staged x_t (K==128)

  const int tid  = threadIdx.x;
  const int lane = tid & 31;
  const int wave = tid >> 5;                  // 0..31 -> N-tile
  const int row0 = blockIdx.x * ROWS_PB;
  constexpr int KOFF = K - HIDDEN;            // offset of h inside A
  const uint32_t a_base = (uint32_t)(uintptr_t)(&a_lds[0][0]);

  // ---- preload B fragments (stay in VGPRs across the whole t-loop) ----
  // B layout (32x16, K x N): lane L holds column n=L&15,
  // K = (L>>4)*8 + 0..7 and 16 + (L>>4)*8 + 0..7 from row-major weights.
  const int ncol = wave * 16 + (lane & 15);
  const int kseg = (lane >> 4) << 3;
  v16bf breg[K / 32];
#pragma unroll
  for (int s = 0; s < K / 32; ++s) {
    v8bf blo = *(const v8bf*)(Wb + (size_t)ncol * K + s * 32 + kseg);
    v8bf bhi = *(const v8bf*)(Wb + (size_t)ncol * K + s * 32 + kseg + 16);
    breg[s] = __builtin_shufflevector(blo, bhi,
        0, 1, 2, 3, 4, 5, 6, 7, 8, 9, 10, 11, 12, 13, 14, 15);
  }

  // ---- elementwise ownership: 2 (batch-row, hidden-unit) pairs / thread ----
  const int b_a = tid >> 7;                   // 0..7
  const int b_b = b_a + 8;                    // 8..15
  const int j_e = tid & 127;
  float c_a = 0.0f, c_b = 0.0f;

  // hoist loop-invariant bias (and layer-0 input-proj weights)
  const float bi_i = bias[j_e];
  const float bi_f = bias[HIDDEN + j_e];
  const float bi_g = bias[2 * HIDDEN + j_e];
  const float bi_o = bias[3 * HIDDEN + j_e];
  float wih[4][5];
  if constexpr (K == 128) {
#pragma unroll
    for (int gq = 0; gq < 4; ++gq)
#pragma unroll
      for (int d = 0; d < 5; ++d)
        wih[gq][d] = W_ih0[(size_t)(gq * HIDDEN + j_e) * 5 + d];
  }

  // zero-init h region of A
  a_lds[b_a][KOFF + j_e] = (bf16_t)0.0f;
  a_lds[b_b][KOFF + j_e] = (bf16_t)0.0f;

  // layer1: async-load x_0 tile (16 rows x 128 bf16 = 4KB contiguous)
  if constexpr (K == 256) {
    if (tid < 256) {
      const bf16_t* gsrc = h1seq_in + (size_t)row0 * HIDDEN;     // t = 0
      uint32_t voff   = (uint32_t)tid * 16u;
      uint32_t ldsoff = a_base + (uint32_t)(tid >> 4) * (uint32_t)(K * 2)
                               + (uint32_t)(tid & 15) * 16u;
      asm volatile("global_load_async_to_lds_b128 %0, %1, %2"
                   :: "v"(ldsoff), "v"(voff), "s"(gsrc) : "memory");
      asm volatile("s_wait_asynccnt 0" ::: "memory");
    }
  }
  __syncthreads();

  const int m     = lane & 15;                // A row this lane feeds
  const int koff  = (lane >> 4) << 3;         // A K-half select
  const int mbase = (lane >> 4) << 3;         // C/D row base

  for (int t = 0; t < TSTEPS; ++t) {
    if constexpr (K == 128) {
      // async-store h_{t-1} (already in LDS, contiguous 4KB) to global
      if (t > 0 && tid < 256) {
        const bf16_t* gdst = h1seq_out + (size_t)(t - 1) * BATCH * HIDDEN
                                       + (size_t)row0 * HIDDEN;
        uint32_t voff   = (uint32_t)tid * 16u;
        uint32_t ldsoff = a_base + (uint32_t)tid * 16u;
        asm volatile("global_store_async_from_lds_b128 %0, %1, %2"
                     :: "v"(voff), "v"(ldsoff), "s"(gdst) : "memory");
      }
      // stage x_t slice (16 rows x 5 f32) into LDS during GEMM phase
      if (tid < 80) {
        int bb = tid / 5, d = tid - bb * 5;
        x_lds[bb][d] = x[((size_t)(row0 + bb) * TSTEPS + t) * 5 + d];
      }
    }

    // ================= GEMM: gates = A x Wb^T =================
    v8f acc = {};
#pragma unroll
    for (int s = 0; s < K / 32; ++s) {
      // A layout (16x32, M x K): lane L row m=L&15;
      // halves 0..7 = K s*32+koff.. , halves 8..15 = +16.
      const bf16_t* ap = &a_lds[m][s * 32 + koff];
      v8bf lo = *(const v8bf*)(ap);
      v8bf hi = *(const v8bf*)(ap + 16);
      v16bf afrag = __builtin_shufflevector(lo, hi,
          0, 1, 2, 3, 4, 5, 6, 7, 8, 9, 10, 11, 12, 13, 14, 15);
      acc = __builtin_amdgcn_wmma_f32_16x16x32_bf16(
          false, afrag, false, breg[s], (short)0, acc, false, false);
    }
    // C/D layout: lane L, reg r -> M = (L>>4)*8 + r, N = L&15
#pragma unroll
    for (int r = 0; r < 8; ++r)
      gates_lds[mbase + r][ncol] = acc[r];

    // h_{t-1} async store must complete before elementwise overwrites h
    if constexpr (K == 128) {
      if (t > 0 && tid < 256)
        asm volatile("s_wait_asynccnt 0" ::: "memory");
    }
    __syncthreads();   // gates + x_lds visible; A reads of step t complete

    // layer1: prefetch x_{t+1} into A (safe: A reads for step t are done)
    if constexpr (K == 256) {
      if (t + 1 < TSTEPS && tid < 256) {
        const bf16_t* gsrc = h1seq_in + (size_t)(t + 1) * BATCH * HIDDEN
                                      + (size_t)row0 * HIDDEN;
        uint32_t voff   = (uint32_t)tid * 16u;
        uint32_t ldsoff = a_base + (uint32_t)(tid >> 4) * (uint32_t)(K * 2)
                                 + (uint32_t)(tid & 15) * 16u;
        asm volatile("global_load_async_to_lds_b128 %0, %1, %2"
                     :: "v"(ldsoff), "v"(voff), "s"(gsrc) : "memory");
      }
    }

    // ================= elementwise gate update =================
#pragma unroll
    for (int p = 0; p < 2; ++p) {
      const int b = (p == 0) ? b_a : b_b;
      const int j = j_e;
      float gi = gates_lds[b][j]              + bi_i;
      float gf = gates_lds[b][HIDDEN + j]     + bi_f;
      float gg = gates_lds[b][2 * HIDDEN + j] + bi_g;
      float go = gates_lds[b][3 * HIDDEN + j] + bi_o;

      if constexpr (K == 128) {
        const float x0 = x_lds[b][0], x1 = x_lds[b][1], x2 = x_lds[b][2],
                    x3 = x_lds[b][3], x4 = x_lds[b][4];
        gi += x0*wih[0][0] + x1*wih[0][1] + x2*wih[0][2] + x3*wih[0][3] + x4*wih[0][4];
        gf += x0*wih[1][0] + x1*wih[1][1] + x2*wih[1][2] + x3*wih[1][3] + x4*wih[1][4];
        gg += x0*wih[2][0] + x1*wih[2][1] + x2*wih[2][2] + x3*wih[2][3] + x4*wih[2][4];
        go += x0*wih[3][0] + x1*wih[3][1] + x2*wih[3][2] + x3*wih[3][3] + x4*wih[3][4];
      }

      const float ig = fsig(gi);
      const float fg = fsig(gf);
      const float g  = ftanh(gg);
      const float og = fsig(go);

      float& c = (p == 0) ? c_a : c_b;
      c = fg * c + ig * g;
      const float h = og * ftanh(c);

      a_lds[b][KOFF + j] = (bf16_t)h;
      if constexpr (K == 256)
        if (t == TSTEPS - 1)
          hfinal[(size_t)(row0 + b) * HIDDEN + j] = h;
    }

    if constexpr (K == 256) {
      if (t + 1 < TSTEPS && tid < 256)
        asm volatile("s_wait_asynccnt 0" ::: "memory");
    }
    __syncthreads();   // h (and x) ready for step t+1
  }

  // layer0: store final h_{T-1}
  if constexpr (K == 128) {
    if (tid < 256) {
      const bf16_t* gdst = h1seq_out + (size_t)(TSTEPS - 1) * BATCH * HIDDEN
                                     + (size_t)row0 * HIDDEN;
      uint32_t voff   = (uint32_t)tid * 16u;
      uint32_t ldsoff = a_base + (uint32_t)tid * 16u;
      asm volatile("global_store_async_from_lds_b128 %0, %1, %2"
                   :: "v"(voff), "v"(ldsoff), "s"(gdst) : "memory");
      asm volatile("s_wait_asynccnt 0" ::: "memory");
    }
  }
}

// ------------------------------------------------------------------
// Final classification head: out[b][c] = h_final[b] . fc_W[c] + fc_b[c]
// ------------------------------------------------------------------
__global__ void fc_kernel(const float* __restrict__ hfinal,
                          const float* __restrict__ fcW,
                          const float* __restrict__ fcb,
                          float* __restrict__ out) {
  int b = blockIdx.x * blockDim.x + threadIdx.x;
  if (b >= BATCH) return;
  const float* h = hfinal + (size_t)b * HIDDEN;
  float a0 = 0.0f, a1 = 0.0f;
#pragma unroll 4
  for (int j = 0; j < HIDDEN; ++j) {
    const float v = h[j];
    a0 += v * fcW[j];
    a1 += v * fcW[HIDDEN + j];
  }
  out[b * 2 + 0] = a0 + fcb[0];
  out[b * 2 + 1] = a1 + fcb[1];
}

extern "C" void kernel_launch(void* const* d_in, const int* in_sizes, int n_in,
                              void* d_out, int out_size, void* d_ws, size_t ws_size,
                              hipStream_t stream) {
  const float* x     = (const float*)d_in[0];
  const float* W_ih0 = (const float*)d_in[1];
  const float* W_hh0 = (const float*)d_in[2];
  const float* b_ih0 = (const float*)d_in[3];
  const float* b_hh0 = (const float*)d_in[4];
  const float* W_ih1 = (const float*)d_in[5];
  const float* W_hh1 = (const float*)d_in[6];
  const float* b_ih1 = (const float*)d_in[7];
  const float* b_hh1 = (const float*)d_in[8];
  const float* fcW   = (const float*)d_in[9];
  const float* fcb   = (const float*)d_in[10];

  char* ws = (char*)d_ws;
  bf16_t* W0     = (bf16_t*)(ws + WS_W0);
  bf16_t* Wc1    = (bf16_t*)(ws + WS_WC1);
  float*  b0     = (float*) (ws + WS_B0);
  float*  b1     = (float*) (ws + WS_B1);
  float*  hfinal = (float*) (ws + WS_HFIN);
  bf16_t* h1seq  = (bf16_t*)(ws + WS_H1SEQ);

  prep_kernel<<<dim3(512), dim3(128), 0, stream>>>(
      W_hh0, W_ih1, W_hh1, b_ih0, b_hh0, b_ih1, b_hh1, W0, Wc1, b0, b1);

  lstm_layer_kernel<128><<<dim3(BATCH / ROWS_PB), dim3(BLK_THR), 0, stream>>>(
      x, W_ih0, W0, b0, nullptr, h1seq, nullptr);

  lstm_layer_kernel<256><<<dim3(BATCH / ROWS_PB), dim3(BLK_THR), 0, stream>>>(
      nullptr, nullptr, Wc1, b1, h1seq, nullptr, hfinal);

  fc_kernel<<<dim3(1), dim3(BATCH), 0, stream>>>(hfinal, fcW, fcb, (float*)d_out);
}
